// Model_47742856463049
// MI455X (gfx1250) — compile-verified, matrix operand order checked by reading, not run on hardware
//
#include <hip/hip_runtime.h>
#include <math.h>

// ---------------------------------------------------------------------------
// CDNA5 (gfx1250) implementation of the CIF + upsample + BiLSTM reference.
// All dense math runs through v_wmma_f32_16x16x32_f16 (f16 in, f32 accum).
// ---------------------------------------------------------------------------

typedef __attribute__((ext_vector_type(16))) _Float16 v16h;
typedef __attribute__((ext_vector_type(8)))  _Float16 v8h;
typedef __attribute__((ext_vector_type(8)))  float    v8f;

union AFrag { v16h v; struct { v8h lo, hi; } p; };

#define THRESH   1.0f
#define TAIL_T   0.45f
#define SMOOTH2  0.25f
#define NOISE_T2 0.01f
#define PEAK_T   0.9999f

__device__ __forceinline__ float fast_rcp(float x) { return __builtin_amdgcn_rcpf(x); }
__device__ __forceinline__ float sigf(float x) { return fast_rcp(1.0f + __expf(-x)); }
// branchless tanh: saturates to +/-1 for large |x| (exp -> inf or 0)
__device__ __forceinline__ float tanh_fast(float x) {
    float e = __expf(2.0f * x);
    return 1.0f - 2.0f * fast_rcp(e + 1.0f);
}

// ---------------------------------------------------------------------------
// Packing kernels: build f16 A/B operands in WMMA-fragment-friendly layouts.
// B swizzle: element (k,n) -> [kk=k>>5][nt=n>>4][lane][e] with
//   lane = 16*((k>>3)&1) + (n&15),  e = 8*((k>>4)&1) + (k&7)
// so a lane's v16h fragment is one contiguous 32B load.
// ---------------------------------------------------------------------------
__global__ void pack_b_kernel(const float* __restrict__ src, _Float16* __restrict__ dst,
                              int K, int N, int kind) {
    size_t total = (size_t)K * N;
    for (size_t id = blockIdx.x * (size_t)blockDim.x + threadIdx.x; id < total;
         id += (size_t)gridDim.x * blockDim.x) {
        int k = (int)(id / N), n = (int)(id % N);
        float v;
        if (kind == 0) {        // conv GEMM B: k = tap*512+ci, n = co; conv_w (co,ci,tap)
            v = src[(size_t)n * 1536 + (size_t)(k & 511) * 3 + (k >> 9)];
        } else if (kind == 1) { // ups GEMM B: k = ci, n = tap*512+co; ups_w[ci,co,tap] (transpose_kernel)
            v = src[(size_t)k * 1536 + (size_t)(n & 511) * 3 + (n >> 9)];
        } else {                // wi/wh transposed: B[k][n] = w[n*512 + k]
            v = src[(size_t)n * 512 + k];
        }
        int lane = (((k >> 3) & 1) << 4) + (n & 15);
        int e    = (((k >> 4) & 1) << 3) + (k & 7);
        size_t off = (((size_t)(k >> 5) * (N >> 4) + (n >> 4)) * 32 + lane) * 16 + e;
        dst[off] = (_Float16)v;
    }
}

// hidden (8,1024,512) f32 -> padded f16 (8,1026,512), zero rows at s=0 and s=1025
__global__ void pad_hidden_kernel(const float* __restrict__ hidden, _Float16* __restrict__ dst) {
    size_t total = (size_t)8 * 1026 * 512;
    for (size_t id = blockIdx.x * (size_t)blockDim.x + threadIdx.x; id < total;
         id += (size_t)gridDim.x * blockDim.x) {
        int b = (int)(id / (1026 * 512));
        int r = (int)(id % (1026 * 512));
        int s = r / 512, c = r % 512;
        float v = (s == 0 || s == 1025) ? 0.0f
                  : hidden[((size_t)b * 1024 + (s - 1)) * 512 + c];
        dst[id] = (_Float16)v;
    }
}

__global__ void zero_f32_kernel(float* p, int n) {
    int i = blockIdx.x * blockDim.x + threadIdx.x;
    if (i < n) p[i] = 0.0f;
}

// ---------------------------------------------------------------------------
// Generic tiled WMMA GEMM. Block = 256 threads = 8 waves (4M x 2N),
// block tile 64M x 128N; wave tile 16M x 64N (A-frag reused over 4 wmma).
// A is row-major f16 with row mapping m -> m*lda + (m/rowDiv)*rowGap
// (lets the conv view padded-hidden rows, with contiguous K=1536 im2col).
// B comes pre-swizzled. Epilogues:
//   MODE 1: conv head  -> relu(acc+bias)*vecw, lane-reduced, atomicAdd outF[m]
//   MODE 2: upsample   -> outH[(3*m + n>>9)*512 + (n&511)] = f16(acc + bias[n&511])
//   MODE 3: xt in C-fragment layout [t][n>>4][n&15][b] (b = m/tdim, t = m%tdim)
//           so the recurrent kernel acc-inits with one b128 load per gate tile.
// ---------------------------------------------------------------------------
template <int MODE>
__global__ __launch_bounds__(256) void gemm_wmma_kernel(
    const _Float16* __restrict__ A, int lda, int rowDiv, int rowGap,
    const _Float16* __restrict__ Bsw, int M, int N, int K,
    const float* __restrict__ bias, const float* __restrict__ vecw,
    float* __restrict__ outF, _Float16* __restrict__ outH, int tdim) {
    int wave = threadIdx.x >> 5;
    int lane = threadIdx.x & 31;
    int hi   = lane >> 4;
    int mw   = wave >> 1, nw = wave & 1;
    int mBase  = blockIdx.x * 64 + mw * 16;
    int ntBase = blockIdx.y * 8 + nw * 4;      // 16-col tiles
    int NT = N >> 4;

    int mRow = mBase + (lane & 15);
    const _Float16* aRow = A + (size_t)mRow * lda + (size_t)(mRow / rowDiv) * rowGap + (hi << 3);
    const _Float16* bPtr = Bsw + ((size_t)ntBase * 32 + lane) * 16;
    size_t bStride = (size_t)NT * 512;

    v8f acc[4] = {v8f{}, v8f{}, v8f{}, v8f{}};
    int kSteps = K >> 5;
    for (int kk = 0; kk < kSteps; ++kk) {
        AFrag af;
        af.p.lo = *(const v8h*)(aRow);
        af.p.hi = *(const v8h*)(aRow + 16);
        aRow += 32;
        __builtin_prefetch(aRow + 32, 0, 3);   // global_prefetch on A stream
#pragma unroll
        for (int j = 0; j < 4; ++j) {
            v16h bf = *(const v16h*)(bPtr + (size_t)j * 512);
            acc[j] = __builtin_amdgcn_wmma_f32_16x16x32_f16(
                false, af.v, false, bf, (short)0, acc[j], false, false);
        }
        bPtr += bStride;
    }

    if (MODE == 1) {
        float part[8];
#pragma unroll
        for (int r = 0; r < 8; ++r) part[r] = 0.0f;
#pragma unroll
        for (int j = 0; j < 4; ++j) {
            int n = ((ntBase + j) << 4) + (lane & 15);
            float bn = bias[n], wn = vecw[n];
#pragma unroll
            for (int r = 0; r < 8; ++r) {
                float v = acc[j][r] + bn;
                v = fmaxf(v, 0.0f);
                part[r] += v * wn;
            }
        }
#pragma unroll
        for (int r = 0; r < 8; ++r) {
            float s = part[r];
            s += __shfl_xor(s, 1);
            s += __shfl_xor(s, 2);
            s += __shfl_xor(s, 4);
            s += __shfl_xor(s, 8);
            if ((lane & 15) == 0) atomicAdd(outF + (mBase + r + (hi << 3)), s);
        }
    } else if (MODE == 2) {
#pragma unroll
        for (int j = 0; j < 4; ++j) {
            int n = ((ntBase + j) << 4) + (lane & 15);
            float bn = bias[n & 511];
#pragma unroll
            for (int r = 0; r < 8; ++r) {
                int m = mBase + r + (hi << 3);
                outH[((size_t)(3 * m + (n >> 9))) * 512 + (n & 511)] = (_Float16)(acc[j][r] + bn);
            }
        }
    } else {  // MODE 3: swizzled xt store for the recurrent kernel
#pragma unroll
        for (int j = 0; j < 4; ++j) {
            int n = ((ntBase + j) << 4) + (lane & 15);
            float bn = bias[n];
#pragma unroll
            for (int r = 0; r < 8; ++r) {
                int m = mBase + r + (hi << 3);
                int b = m / tdim, t = m % tdim;
                outH[(((size_t)t * NT + (n >> 4)) * 16 + (n & 15)) * 8 + b] =
                    (_Float16)(acc[j][r] + bn);
            }
        }
    }
}

// ---------------------------------------------------------------------------
// alphas head finalize: sigmoid + mask, build alphas_t (b,1025) with tail.
// ---------------------------------------------------------------------------
__global__ void alphas_finalize_kernel(const float* __restrict__ alphaPre,
                                       const float* __restrict__ cifB,
                                       const float* __restrict__ mask,
                                       float* __restrict__ alphasOut,
                                       float* __restrict__ alphasT) {
    int id = blockIdx.x * blockDim.x + threadIdx.x;
    if (id >= 8 * 1025) return;
    int b = id / 1025, s = id % 1025;
    float a = 0.0f;
    if (s < 1024) {
        float sg = sigf(alphaPre[b * 1024 + s] + cifB[0]);
        a = fmaxf(sg, 0.0f) * mask[b * 1024 + s];   // SMOOTH=1, NOISE_T=0
        alphasOut[b * 1024 + s] = a;
    }
    float m1 = (s < 1024) ? mask[b * 1024 + s] : 0.0f;
    float m2 = (s == 0) ? 1.0f : mask[b * 1024 + s - 1];
    alphasT[b * 1025 + s] = a + (m2 - m1) * TAIL_T;
}

// ---------------------------------------------------------------------------
// CIF scan: one block per batch, thread d owns one feature. Also emits
// token_num = floor(sum(alphas_t)).
// ---------------------------------------------------------------------------
__global__ __launch_bounds__(512) void cif_kernel(const float* __restrict__ hidden,
                                                  const float* __restrict__ alphasT,
                                                  float* __restrict__ embeds,
                                                  float* __restrict__ tokenNum) {
    int b = blockIdx.x, d = threadIdx.x;
    for (int i = d; i < 1025 * 512; i += 512) embeds[(size_t)b * 1025 * 512 + i] = 0.0f;
    __shared__ float red[512];
    float s = 0.0f;
    for (int i = d; i < 1025; i += 512) s += alphasT[b * 1025 + i];
    red[d] = s;
    __syncthreads();
    for (int off = 256; off > 0; off >>= 1) {
        if (d < off) red[d] += red[d + off];
        __syncthreads();
    }
    if (d == 0) tokenNum[b] = floorf(red[0]);

    float integrate = 0.0f, frame = 0.0f;
    int pos = 0;
    for (int t = 0; t < 1025; ++t) {
        float alpha = alphasT[b * 1025 + t];
        float h = (t < 1024) ? hidden[((size_t)b * 1024 + t) * 512 + d] : 0.0f;
        float distComp = 1.0f - integrate;
        integrate += alpha;
        bool fire = (integrate >= THRESH);
        float cur = fire ? distComp : alpha;
        float frameOut = frame + cur * h;
        if (fire) {
            if (pos < 1025) embeds[((size_t)b * 1025 + pos) * 512 + d] = frameOut;
            pos++;
            frame = (alpha - cur) * h;
            integrate -= 1.0f;
        } else {
            frame = frameOut;
        }
    }
}

// ---------------------------------------------------------------------------
// Persistent LSTM recurrence (one direction). 1024 threads = 32 waves.
// h (16x512 f16) lives in LDS; wave w owns gate columns d in [16w,16w+16):
// its 4 WMMA tiles are the i/f/g/o columns for those d — so the nonlinearity
// and c/h update are fully lane-local (no cross-wave gate exchange).
// Wh fragments are re-read from L2 each step (2 MB, L2-resident). LICM is
// blinded with an opaque *scalar* zero offset (not a pointer round-trip, so
// address-space inference keeps the loads as global_load_b128, not flat).
// xt arrives pre-swizzled in C-fragment layout: one b128 load per gate tile.
// Fuses the blstm @ cif_out2_w dot per step (saves storing hf/hb entirely).
// ---------------------------------------------------------------------------
__global__ __launch_bounds__(1024) void lstm_dir_kernel(
    const _Float16* __restrict__ xtSw,   // [t][nt=128][lane16][r=batch] f16
    const _Float16* __restrict__ WhSw,   // swizzled (K=512, N=2048)
    const float* __restrict__ w2,        // 512 entries for this direction
    float* __restrict__ usPre,           // (8*T) partial dot output
    int T, int dir) {
    __shared__ _Float16 hbuf[16 * 512];
    __shared__ float usAcc[16];
    int tid = threadIdx.x;
    int wave = tid >> 5, lane = tid & 31, hi = lane >> 4;
    int d = (wave << 4) + (lane & 15);
    float w2v = w2[d];

    for (int i = tid; i < 16 * 512; i += 1024) hbuf[i] = (_Float16)0;
    if (tid < 16) usAcc[tid] = 0.0f;
    __syncthreads();

    v8f cfrag = v8f{};
    for (int step = 0; step < T; ++step) {
        int t = dir ? (T - 1 - step) : step;

        // Opaque scalar 0: addresses formally vary per step, so the 64 weight
        // fragment loads cannot be hoisted out of the time loop (which would
        // need 512 VGPRs and spill), but the pointer chain stays intact and
        // the loads remain global (not flat).
        unsigned hoistGuard = 0;
        asm volatile("" : "+s"(hoistGuard));
        const _Float16* wpl = WhSw + ((size_t)wave * 32 + lane) * 16 + hoistGuard;

        // acc init from swizzled xt: 4 coalesced b128 loads (lanes 0-15),
        // rows 8-15 of the padded M=16 tile are zero.
        v8h xv[4];
#pragma unroll
        for (int g = 0; g < 4; ++g) xv[g] = v8h{};
        if (hi == 0) {
            const _Float16* xp = xtSw + (size_t)t * (128 * 16 * 8);
#pragma unroll
            for (int g = 0; g < 4; ++g) {
                int nt = g * 32 + wave;
                xv[g] = *(const v8h*)(xp + ((size_t)nt * 16 + (lane & 15)) * 8);
            }
        }
        v8f acc[4];
#pragma unroll
        for (int g = 0; g < 4; ++g)
#pragma unroll
            for (int r = 0; r < 8; ++r) acc[g][r] = (float)xv[g][r];

        const _Float16* aPtr = hbuf + (lane & 15) * 512 + (hi << 3);
#pragma unroll 1
        for (int kk = 0; kk < 16; ++kk) {
            AFrag af;
            af.p.lo = *(const v8h*)(aPtr);        // ds_read_b128
            af.p.hi = *(const v8h*)(aPtr + 16);
            aPtr += 32;
#pragma unroll
            for (int g = 0; g < 4; ++g) {
                v16h bf = *(const v16h*)(wpl + (size_t)kk * 65536 + (size_t)g * 16384);
                acc[g] = __builtin_amdgcn_wmma_f32_16x16x32_f16(
                    false, af.v, false, bf, (short)0, acc[g], false, false);
            }
        }

        // Branchless gate math on the serial critical path (v_exp + v_rcp,
        // no libm range-check branches).
        v8f hNew;
#pragma unroll
        for (int r = 0; r < 8; ++r) {
            float iv = sigf(acc[0][r]);
            float fv = sigf(acc[1][r]);
            float gv = tanh_fast(acc[2][r]);
            float ov = sigf(acc[3][r]);
            float c = fv * cfrag[r] + iv * gv;
            cfrag[r] = c;
            hNew[r] = ov * tanh_fast(c);
        }
        __syncthreads();                        // all waves done reading hbuf
#pragma unroll
        for (int r = 0; r < 8; ++r) {
            if (hi == 0) hbuf[(r)*512 + d] = (_Float16)hNew[r];   // rows 8..15 stay 0
            float contrib = (hi == 0) ? hNew[r] * w2v : 0.0f;
            contrib += __shfl_xor(contrib, 1);
            contrib += __shfl_xor(contrib, 2);
            contrib += __shfl_xor(contrib, 4);
            contrib += __shfl_xor(contrib, 8);
            if (lane == 0) atomicAdd(&usAcc[r], contrib);         // ds_add_f32
        }
        __syncthreads();                        // hbuf + usAcc complete
        if (tid < 8) {
            usPre[(size_t)tid * T + t] = usAcc[tid];
            usAcc[tid] = 0.0f;
        }
    }
}

// ---------------------------------------------------------------------------
// us_alphas finalize + rescale + us_peaks scan. One block per batch.
// ---------------------------------------------------------------------------
__global__ __launch_bounds__(256) void us_finalize_kernel(
    const float* __restrict__ usPre,   // [2][8][T]
    const float* __restrict__ b2, const float* __restrict__ mask,
    const float* __restrict__ tokenNum,
    float* __restrict__ usAlphas, float* __restrict__ usPeaks) {
    const int T = 3072;
    int b = blockIdx.x, tid = threadIdx.x;
    __shared__ float red[256];
    float s = 0.0f;
    for (int t = tid; t < T; t += 256) {
        float pre = usPre[(size_t)b * T + t] + usPre[(size_t)8 * T + (size_t)b * T + t] + b2[0];
        float a = fmaxf(sigf(pre) * SMOOTH2 - NOISE_T2, 0.0f) * mask[b * 1024 + t / 3];
        usAlphas[(size_t)b * T + t] = a;
        s += a;
    }
    red[tid] = s;
    __syncthreads();
    for (int off = 128; off > 0; off >>= 1) {
        if (tid < off) red[tid] += red[tid + off];
        __syncthreads();
    }
    float scale = tokenNum[b] / red[0];
    for (int t = tid; t < T; t += 256) usAlphas[(size_t)b * T + t] *= scale;
    __syncthreads();
    if (tid == 0) {
        float integ = 0.0f;
        for (int t = 0; t < T; ++t) {
            integ += usAlphas[(size_t)b * T + t];
            usPeaks[(size_t)b * T + t] = integ;
            if (integ >= PEAK_T) integ -= PEAK_T;
        }
    }
}

// ---------------------------------------------------------------------------
// Host launcher
// ---------------------------------------------------------------------------
extern "C" void kernel_launch(void* const* d_in, const int* in_sizes, int n_in,
                              void* d_out, int out_size, void* d_ws, size_t ws_size,
                              hipStream_t stream) {
    const float* hidden = (const float*)d_in[0];
    const float* mask   = (const float*)d_in[1];
    const float* conv_w = (const float*)d_in[2];
    const float* conv_b = (const float*)d_in[3];
    const float* cif_w  = (const float*)d_in[4];
    const float* cif_b  = (const float*)d_in[5];
    const float* ups_w  = (const float*)d_in[6];
    const float* ups_b  = (const float*)d_in[7];
    const float* wi_f   = (const float*)d_in[8];
    const float* wh_f   = (const float*)d_in[9];
    const float* b_f    = (const float*)d_in[10];
    const float* wi_b   = (const float*)d_in[11];
    const float* wh_b   = (const float*)d_in[12];
    const float* b_b    = (const float*)d_in[13];
    const float* w2     = (const float*)d_in[14];
    const float* b2     = (const float*)d_in[15];
    float* out = (float*)d_out;

    // d_out layout (floats): embeds (8,1025,512) | token_num (8) | alphas (8,1024)
    //                        | us_alphas (8,3072) | us_peaks (8,3072)
    const size_t OFF_TOK = (size_t)8 * 1025 * 512;
    const size_t OFF_AL  = OFF_TOK + 8;
    const size_t OFF_UA  = OFF_AL + (size_t)8 * 1024;
    const size_t OFF_UP  = OFF_UA + (size_t)8 * 3072;

    // Workspace carve (~146 MB).
    char* wsp = (char*)d_ws;
    auto carve = [&](size_t bytes) -> void* {
        void* p = (void*)wsp;
        wsp += (bytes + 255) & ~(size_t)255;
        return p;
    };
    _Float16* hidPad = (_Float16*)carve((size_t)8 * 1026 * 512 * 2);
    _Float16* convB  = (_Float16*)carve((size_t)1536 * 512 * 2);
    _Float16* upsB   = (_Float16*)carve((size_t)512 * 1536 * 2);
    _Float16* wiTf   = (_Float16*)carve((size_t)512 * 2048 * 2);
    _Float16* whTf   = (_Float16*)carve((size_t)512 * 2048 * 2);
    _Float16* wiTb   = (_Float16*)carve((size_t)512 * 2048 * 2);
    _Float16* whTb   = (_Float16*)carve((size_t)512 * 2048 * 2);
    _Float16* upsH   = (_Float16*)carve((size_t)8 * 3072 * 512 * 2);
    _Float16* xtSw   = (_Float16*)carve((size_t)3072 * 128 * 16 * 8 * 2);  // reused fwd/bwd
    float* alphaPre  = (float*)carve((size_t)8192 * 4);
    float* alphasT   = (float*)carve((size_t)8 * 1025 * 4);
    float* usPre     = (float*)carve((size_t)2 * 8 * 3072 * 4);

    const int BIG = 1 << 30;

    zero_f32_kernel<<<32, 256, 0, stream>>>(alphaPre, 8192);
    pad_hidden_kernel<<<8192, 256, 0, stream>>>(hidden, hidPad);
    pack_b_kernel<<<3072, 256, 0, stream>>>(conv_w, convB, 1536, 512, 0);
    pack_b_kernel<<<3072, 256, 0, stream>>>(ups_w, upsB, 512, 1536, 1);
    pack_b_kernel<<<4096, 256, 0, stream>>>(wi_f, wiTf, 512, 2048, 2);
    pack_b_kernel<<<4096, 256, 0, stream>>>(wh_f, whTf, 512, 2048, 2);
    pack_b_kernel<<<4096, 256, 0, stream>>>(wi_b, wiTb, 512, 2048, 2);
    pack_b_kernel<<<4096, 256, 0, stream>>>(wh_b, whTb, 512, 2048, 2);

    // Conv (im2col via padded rows: K=1536 contiguous, lda=512) fused with alpha head.
    gemm_wmma_kernel<1><<<dim3(128, 4), 256, 0, stream>>>(
        hidPad, 512, 1024, 1024, convB, 8192, 512, 1536, conv_b, cif_w, alphaPre, nullptr, 0);
    alphas_finalize_kernel<<<(8 * 1025 + 255) / 256, 256, 0, stream>>>(
        alphaPre, cif_b, mask, out + OFF_AL, alphasT);
    cif_kernel<<<8, 512, 0, stream>>>(hidden, alphasT, out, out + OFF_TOK);

    // Upsample (stride==kernel => scatter epilogue), output directly in f16 for LSTM A.
    gemm_wmma_kernel<2><<<dim3(128, 12), 256, 0, stream>>>(
        hidPad + 512, 512, 1024, 1024, upsB, 8192, 1536, 512, ups_b, nullptr, nullptr, upsH, 0);

    // Forward LSTM: parallel input projection then persistent recurrence.
    gemm_wmma_kernel<3><<<dim3(384, 16), 256, 0, stream>>>(
        upsH, 512, BIG, 0, wiTf, 24576, 2048, 512, b_f, nullptr, nullptr, xtSw, 3072);
    lstm_dir_kernel<<<1, 1024, 0, stream>>>(xtSw, whTf, w2, usPre, 3072, 0);

    // Backward LSTM (xtSw buffer reused).
    gemm_wmma_kernel<3><<<dim3(384, 16), 256, 0, stream>>>(
        upsH, 512, BIG, 0, wiTb, 24576, 2048, 512, b_b, nullptr, nullptr, xtSw, 3072);
    lstm_dir_kernel<<<1, 1024, 0, stream>>>(xtSw, whTb, w2 + 512, usPre + (size_t)8 * 3072, 3072, 1);

    us_finalize_kernel<<<8, 256, 0, stream>>>(usPre, b2, mask, out + OFF_TOK,
                                              out + OFF_UA, out + OFF_UP);
}
